// SlotAttention_13280038879906
// MI455X (gfx1250) — compile-verified, matrix operand order checked by reading, not run on hardware
//
#include <hip/hip_runtime.h>
#include <hip/hip_bf16.h>

// ---------------------------------------------------------------------------
// SlotAttention for MI455X / gfx1250 (wave32, WMMA f16 -> f32 accum)
// ---------------------------------------------------------------------------
constexpr int E   = 262144;
constexpr int NN  = 16384;
constexpr int H   = 8;
constexpr int D   = 64;
constexpr int HD  = 512;      // H*D
constexpr int HID = 256;
constexpr int EH  = E * H;    // 2097152

typedef __attribute__((ext_vector_type(16))) _Float16 v16h;
typedef __attribute__((ext_vector_type(8)))  _Float16 v8h;
typedef __attribute__((ext_vector_type(8)))  float    v8f;
typedef __attribute__((ext_vector_type(4)))  float    v4f;

// ---- fragment helpers ------------------------------------------------------
// 16-bit A/B frag: lane (idx=lane&15, hi=lane>>4) holds two contiguous 16-byte
// chunks of row `idx`: bytes [hi*16,+16) and [32+hi*16,+16) of the 64-byte
// K=32 segment (CDNA5 ISA 7.12.2 layout). Works for global or LDS pointers.
__device__ __forceinline__ v16h load_frag(const _Float16* row, int kByte, int hi) {
  const char* base = (const char*)row + kByte + hi * 16;
  v8h lo8 = *(const v8h*)(base);
  v8h hi8 = *(const v8h*)(base + 32);
  v16h r;
#pragma unroll
  for (int i = 0; i < 8; ++i) { r[i] = lo8[i]; r[i + 8] = hi8[i]; }
  return r;
}

__device__ __forceinline__ v8f wmma32(v16h a, v16h b, v8f c) {
  return __builtin_amdgcn_wmma_f32_16x16x32_f16(false, a, false, b, (short)0, c,
                                                false, false);
}

__device__ __forceinline__ float sigmoidf_(float x) {
  return 1.f / (1.f + __expf(-x));
}

// order-preserving float<->uint for atomicMax on floats
__device__ __forceinline__ unsigned f2ord(float f) {
  unsigned u = __float_as_uint(f);
  return (u & 0x80000000u) ? ~u : (u | 0x80000000u);
}
__device__ __forceinline__ float ord2f(unsigned u) {
  return __uint_as_float((u & 0x80000000u) ? (u & 0x7fffffffu) : ~u);
}

// ---------------------------------------------------------------------------
// one-time kernels
// ---------------------------------------------------------------------------
__global__ void cvt_f16_kernel(const float* __restrict__ src,
                               _Float16* __restrict__ dst, int n) {
  int i = blockIdx.x * blockDim.x + threadIdx.x;
  if (i < n) dst[i] = (_Float16)src[i];
}

// weff[h,d] = sum_d' atten[h,d'] * Wq[d',d]   (folds the q projection away)
__global__ void weff_kernel(const float* __restrict__ atten,
                            const float* __restrict__ Wq,
                            float* __restrict__ weff) {
  int t = threadIdx.x;               // 512 threads
  int h = t >> 6, d = t & 63;
  float s = 0.f;
  for (int dp = 0; dp < 64; ++dp) s += atten[h * 64 + dp] * Wq[dp * 64 + d];
  weff[h * 64 + d] = s;
}

// slots = mu + exp(log_sigma) * norm_dist ; write f32 + f16 copies
__global__ void slots_init_kernel(const float* __restrict__ nd,
                                  const float* __restrict__ mu,
                                  const float* __restrict__ ls,
                                  float* __restrict__ s32,
                                  _Float16* __restrict__ s16) {
  int rid = blockIdx.x * blockDim.x + threadIdx.x;   // (e*H + h)
  if (rid >= EH) return;
  size_t base = (size_t)rid * 64;
#pragma unroll
  for (int dd = 0; dd < 64; dd += 8) {
    v4f v0 = *(const v4f*)(nd + base + dd);
    v4f v1 = *(const v4f*)(nd + base + dd + 4);
    v4f o0, o1; v8h h8;
#pragma unroll
    for (int j = 0; j < 4; ++j) {
      o0[j] = mu[dd + j]     + __expf(ls[dd + j])     * v0[j];
      o1[j] = mu[dd + 4 + j] + __expf(ls[dd + 4 + j]) * v1[j];
      h8[j] = (_Float16)o0[j]; h8[4 + j] = (_Float16)o1[j];
    }
    *(v4f*)(s32 + base + dd)     = o0;
    *(v4f*)(s32 + base + dd + 4) = o1;
    *(v8h*)(s16 + base + dd)     = h8;
  }
}

// k = (x @ Wk.T) * D^-0.5 ; Wk staged in LDS, one thread per edge
__global__ __launch_bounds__(256) void kproj_kernel(const float* __restrict__ x,
                                                    const float* __restrict__ Wk,
                                                    float* __restrict__ kout) {
  __shared__ float wk[64 * 128];
  for (int i = threadIdx.x; i < 64 * 128; i += 256) wk[i] = Wk[i];
  __syncthreads();
  int e = blockIdx.x * blockDim.x + threadIdx.x;
  if (e >= E) return;
  const float* xr = x + (size_t)e * 128;
  float xv[128];
#pragma unroll
  for (int i = 0; i < 128; i += 4) {
    v4f v = *(const v4f*)(xr + i);
    xv[i] = v[0]; xv[i + 1] = v[1]; xv[i + 2] = v[2]; xv[i + 3] = v[3];
  }
  for (int d = 0; d < 64; ++d) {
    float s = 0.f;
#pragma unroll
    for (int i = 0; i < 128; ++i) s += xv[i] * wk[d * 128 + i];
    kout[(size_t)e * 64 + d] = s * 0.125f;
  }
}

// ---------------------------------------------------------------------------
// per-iteration kernels
// ---------------------------------------------------------------------------
__global__ void seg_init_kernel(unsigned* __restrict__ segmax,
                                float* __restrict__ segsum) {
  int i = blockIdx.x * blockDim.x + threadIdx.x;
  if (i < NN * H) { segmax[i] = 0u; segsum[i] = 0.f; }
}

// LN over D, dot with weff[h], leaky_relu, atomic seg-max
__global__ void ln_alpha_kernel(const float* __restrict__ slots,
                                const float* __restrict__ weff,
                                const float* __restrict__ g,
                                const float* __restrict__ b,
                                const int* __restrict__ eidx,
                                float* __restrict__ alpha,
                                unsigned* __restrict__ segmax) {
  int rid = blockIdx.x * blockDim.x + threadIdx.x;
  if (rid >= EH) return;
  int h = rid & 7, e = rid >> 3;
  const float* xr = slots + (size_t)rid * 64;
  const float* we = weff + h * 64;
  float xv[64]; float sum = 0.f, sq = 0.f;
#pragma unroll
  for (int i = 0; i < 64; i += 4) {
    v4f v = *(const v4f*)(xr + i);
    xv[i] = v[0]; xv[i + 1] = v[1]; xv[i + 2] = v[2]; xv[i + 3] = v[3];
    sum += v[0] + v[1] + v[2] + v[3];
    sq  += v[0] * v[0] + v[1] * v[1] + v[2] * v[2] + v[3] * v[3];
  }
  float mean = sum * (1.f / 64.f);
  float inv  = rsqrtf(sq * (1.f / 64.f) - mean * mean + 1e-5f);
  float dot = 0.f;
#pragma unroll
  for (int d = 0; d < 64; ++d) {
    float s = (xv[d] - mean) * inv * g[d] + b[d];
    dot += s * we[d];
  }
  float a = dot * 0.125f;
  a = a > 0.f ? a : 0.2f * a;                 // leaky_relu(0.2)
  alpha[rid] = a;
  int node = eidx[e];
  atomicMax(&segmax[node * 8 + h], f2ord(a));
}

__global__ void exp_sum_kernel(float* __restrict__ alpha,
                               const int* __restrict__ eidx,
                               const unsigned* __restrict__ segmax,
                               float* __restrict__ segsum) {
  int rid = blockIdx.x * blockDim.x + threadIdx.x;
  if (rid >= EH) return;
  int h = rid & 7, e = rid >> 3;
  int node = eidx[e];
  float m = ord2f(segmax[node * 8 + h]);
  float ev = __expf(alpha[rid] - m);
  alpha[rid] = ev;
  atomicAdd(&segsum[node * 8 + h], ev);
}

// updates[e, h*64+d] = k[e,d] * softmax_alpha   (written directly as f16)
__global__ void updates_kernel(const float* __restrict__ alpha,
                               const int* __restrict__ eidx,
                               const float* __restrict__ segsum,
                               const float* __restrict__ kbuf,
                               _Float16* __restrict__ upd) {
  int rid = blockIdx.x * blockDim.x + threadIdx.x;
  if (rid >= EH) return;
  int h = rid & 7, e = rid >> 3;
  int node = eidx[e];
  float w = alpha[rid] / (segsum[node * 8 + h] + 1e-16f);
  const float* kr = kbuf + (size_t)e * 64;
  _Float16* o = upd + (size_t)e * 512 + (size_t)h * 64;
#pragma unroll
  for (int dd = 0; dd < 64; dd += 8) {
    v4f x0 = *(const v4f*)(kr + dd);
    v4f x1 = *(const v4f*)(kr + dd + 4);
    v8h pk;
#pragma unroll
    for (int j = 0; j < 4; ++j) {
      pk[j] = (_Float16)(x0[j] * w); pk[4 + j] = (_Float16)(x1[j] * w);
    }
    *(v8h*)(o + dd) = pk;
  }
}

// ---------------------------------------------------------------------------
// GRU: two fused [E,512]x[512,1536] GEMMs + gate math.
// WG = 8 waves; wave = 16 rows x 16 cols; 6 WMMA accumulators (gates).
// A/B frags loaded as 2x b128 directly from global (layout-matched chunks).
// ---------------------------------------------------------------------------
__global__ __launch_bounds__(256) void gru_wmma_kernel(
    const _Float16* __restrict__ upd16,     // [E,512]
    const _Float16* __restrict__ slo16,     // [E,512]
    const float*    __restrict__ slotsPrev, // [E,512]
    const _Float16* __restrict__ wih,       // [1536,512]
    const _Float16* __restrict__ whh,       // [1536,512]
    const float* __restrict__ bih, const float* __restrict__ bhh,
    float* __restrict__ slotsOut)           // [E,512]
{
  const int lane = threadIdx.x & 31;
  const int wave = threadIdx.x >> 5;
  const int lo = lane & 15, hi = lane >> 4;
  const int c0 = blockIdx.x * 16;                     // n-tile (fast-varying)
  const int rowBase = (blockIdx.y * 8 + wave) * 16;

  const _Float16* aU = upd16 + (size_t)(rowBase + lo) * 512;
  const _Float16* aS = slo16 + (size_t)(rowBase + lo) * 512;
  const int wr = c0 + lo;
  const _Float16* bRi = wih + (size_t)(wr)          * 512;
  const _Float16* bZi = wih + (size_t)(wr + 512)    * 512;
  const _Float16* bNi = wih + (size_t)(wr + 1024)   * 512;
  const _Float16* bRh = whh + (size_t)(wr)          * 512;
  const _Float16* bZh = whh + (size_t)(wr + 512)    * 512;
  const _Float16* bNh = whh + (size_t)(wr + 1024)   * 512;

  v8f aR = {}, aZ = {}, aN = {}, hR = {}, hZ = {}, hN = {};
#pragma unroll 4
  for (int t = 0; t < 16; ++t) {
    const int kb = t * 64;
    if (t < 15) {                       // prefetch next A chunks
      __builtin_prefetch((const void*)((const char*)aU + kb + 64), 0, 1);
      __builtin_prefetch((const void*)((const char*)aS + kb + 64), 0, 1);
    }
    v16h au = load_frag(aU, kb, hi);
    v16h as = load_frag(aS, kb, hi);
    aR = wmma32(au, load_frag(bRi, kb, hi), aR);
    aZ = wmma32(au, load_frag(bZi, kb, hi), aZ);
    aN = wmma32(au, load_frag(bNi, kb, hi), aN);
    hR = wmma32(as, load_frag(bRh, kb, hi), hR);
    hZ = wmma32(as, load_frag(bZh, kb, hi), hZ);
    hN = wmma32(as, load_frag(bNh, kb, hi), hN);
  }
  const int c = c0 + lo;
  const float biR = bih[c], biZ = bih[c + 512], biN = bih[c + 1024];
  const float bhR = bhh[c], bhZ = bhh[c + 512], bhN = bhh[c + 1024];
#pragma unroll
  for (int j = 0; j < 8; ++j) {
    const int row = rowBase + hi * 8 + j;           // C-layout row
    const size_t idx = (size_t)row * 512 + c;
    float r  = sigmoidf_(aR[j] + biR + hR[j] + bhR);
    float z  = sigmoidf_(aZ[j] + biZ + hZ[j] + bhZ);
    float nn = tanhf(aN[j] + biN + r * (hN[j] + bhN));
    float hp = slotsPrev[idx];
    slotsOut[idx] = (1.f - z) * nn + z * hp;
  }
}

// ---------------------------------------------------------------------------
// MLP: per (e,h) row: LN -> fc1(64->256) -> relu -> fc2(256->64) -> +residual.
// WG = 4 waves x 16 rows. fc1 output re-laid-out through per-wave LDS tile.
// ---------------------------------------------------------------------------
__global__ __launch_bounds__(128) void mlp_wmma_kernel(
    const float* __restrict__ sin,      // [EH,64] post-GRU slots
    const float* __restrict__ g, const float* __restrict__ b,
    const _Float16* __restrict__ w1,    // [256,64]
    const float* __restrict__ b1,
    const _Float16* __restrict__ w2,    // [64,256]
    const float* __restrict__ b2,
    float* __restrict__ out32,          // [EH,64]
    _Float16* __restrict__ out16)       // [EH,64]
{
  __shared__ _Float16 y1[4][16][256];   // 32 KB, per-wave private regions
  const int lane = threadIdx.x & 31;
  const int wave = threadIdx.x >> 5;
  const int lo = lane & 15, hi = lane >> 4;
  const int rowBase = (blockIdx.x * 4 + wave) * 16;

  // --- LN: each lane handles row rowBase+lo (lanes 16..31 mirror for hi=1)
  const float* xr = sin + (size_t)(rowBase + lo) * 64;
  float sum = 0.f, sq = 0.f;
#pragma unroll
  for (int i = 0; i < 64; i += 4) {
    v4f v = *(const v4f*)(xr + i);
    sum += v[0] + v[1] + v[2] + v[3];
    sq  += v[0] * v[0] + v[1] * v[1] + v[2] * v[2] + v[3] * v[3];
  }
  float mean = sum * (1.f / 64.f);
  float inv  = rsqrtf(sq * (1.f / 64.f) - mean * mean + 1e-5f);

  // pack LN'd values straight into the two K=32 A-fragments
  const int o0 = hi * 8, o1 = 16 + hi * 8, o2 = 32 + hi * 8, o3 = 48 + hi * 8;
  v16h a0, a1;
#pragma unroll
  for (int i = 0; i < 8; ++i) {
    a0[i]     = (_Float16)((xr[o0 + i] - mean) * inv * g[o0 + i] + b[o0 + i]);
    a0[8 + i] = (_Float16)((xr[o1 + i] - mean) * inv * g[o1 + i] + b[o1 + i]);
    a1[i]     = (_Float16)((xr[o2 + i] - mean) * inv * g[o2 + i] + b[o2 + i]);
    a1[8 + i] = (_Float16)((xr[o3 + i] - mean) * inv * g[o3 + i] + b[o3 + i]);
  }

  // --- GEMM1 (K=64) + bias + relu -> LDS, 16 column fragments
  for (int f = 0; f < 16; ++f) {
    const _Float16* br = w1 + (size_t)(f * 16 + lo) * 64;
    v8f acc = {};
    acc = wmma32(a0, load_frag(br, 0, hi), acc);
    acc = wmma32(a1, load_frag(br, 64, hi), acc);
    const float bb = b1[f * 16 + lo];
#pragma unroll
    for (int j = 0; j < 8; ++j)
      y1[wave][hi * 8 + j][f * 16 + lo] = (_Float16)fmaxf(acc[j] + bb, 0.f);
  }
  // per-wave LDS region, in-order DS pipeline: no workgroup barrier required

  // --- GEMM2 (K=256, N=64): A frags re-read from LDS
  const _Float16* yrow = &y1[wave][lo][0];
  v8f o[4] = {};
#pragma unroll
  for (int t = 0; t < 8; ++t) {
    v16h ay = load_frag(yrow, t * 64, hi);
#pragma unroll
    for (int f = 0; f < 4; ++f)
      o[f] = wmma32(ay, load_frag(w2 + (size_t)(f * 16 + lo) * 256, t * 64, hi),
                    o[f]);
  }

  // --- epilogue: residual add, dual-precision writeback
#pragma unroll
  for (int f = 0; f < 4; ++f) {
    const int c = f * 16 + lo;
    const float bb = b2[c];
#pragma unroll
    for (int j = 0; j < 8; ++j) {
      const int row = rowBase + hi * 8 + j;
      const size_t idx = (size_t)row * 64 + c;
      float v = sin[idx] + o[f][j] + bb;
      out32[idx] = v;
      out16[idx] = (_Float16)v;
    }
  }
}

// ---------------------------------------------------------------------------
extern "C" void kernel_launch(void* const* d_in, const int* in_sizes, int n_in,
                              void* d_out, int out_size, void* d_ws,
                              size_t ws_size, hipStream_t stream) {
  (void)in_sizes; (void)n_in; (void)out_size; (void)ws_size;
  const float* x      = (const float*)d_in[0];
  const int*   eidx   = (const int*)d_in[1];
  // d_in[2] = size_i (compile-time NN)
  const float* nd     = (const float*)d_in[3];
  const float* mu     = (const float*)d_in[4];
  const float* logsig = (const float*)d_in[5];
  const float* atten  = (const float*)d_in[6];
  const float* lnS_g  = (const float*)d_in[7];
  const float* lnS_b  = (const float*)d_in[8];
  const float* lnM_g  = (const float*)d_in[9];
  const float* lnM_b  = (const float*)d_in[10];
  const float* Wq     = (const float*)d_in[11];
  const float* Wk     = (const float*)d_in[12];
  const float* wih    = (const float*)d_in[13];
  const float* whh    = (const float*)d_in[14];
  const float* bih    = (const float*)d_in[15];
  const float* bhh    = (const float*)d_in[16];
  const float* fc1w   = (const float*)d_in[17];
  const float* fc1b   = (const float*)d_in[18];
  const float* fc2w   = (const float*)d_in[19];
  const float* fc2b   = (const float*)d_in[20];

  char* p = (char*)d_ws;
  auto alloc = [&](size_t bytes) -> char* {
    char* r = p; p += (bytes + 255) & ~(size_t)255; return r;
  };
  float*    slots32 = (float*)   alloc((size_t)E * HD * 4);
  _Float16* slots16 = (_Float16*)alloc((size_t)E * HD * 2);
  float*    stmp    = (float*)   alloc((size_t)E * HD * 4);
  _Float16* upd16   = (_Float16*)alloc((size_t)E * HD * 2);
  float*    kbuf    = (float*)   alloc((size_t)E * D * 4);
  float*    alpha   = (float*)   alloc((size_t)EH * 4);
  unsigned* segmax  = (unsigned*)alloc((size_t)NN * H * 4);
  float*    segsum  = (float*)   alloc((size_t)NN * H * 4);
  _Float16* wih16   = (_Float16*)alloc((size_t)3 * HD * HD * 2);
  _Float16* whh16   = (_Float16*)alloc((size_t)3 * HD * HD * 2);
  _Float16* w1_16   = (_Float16*)alloc((size_t)HID * D * 2);
  _Float16* w2_16   = (_Float16*)alloc((size_t)D * HID * 2);
  float*    weff    = (float*)   alloc((size_t)H * D * 4);

  const int nW = 3 * HD * HD;           // 786432
  cvt_f16_kernel<<<(nW + 255) / 256, 256, 0, stream>>>(wih, wih16, nW);
  cvt_f16_kernel<<<(nW + 255) / 256, 256, 0, stream>>>(whh, whh16, nW);
  cvt_f16_kernel<<<(HID * D + 255) / 256, 256, 0, stream>>>(fc1w, w1_16, HID * D);
  cvt_f16_kernel<<<(D * HID + 255) / 256, 256, 0, stream>>>(fc2w, w2_16, D * HID);
  weff_kernel<<<1, 512, 0, stream>>>(atten, Wq, weff);
  slots_init_kernel<<<EH / 256, 256, 0, stream>>>(nd, mu, logsig, slots32, slots16);
  kproj_kernel<<<E / 256, 256, 0, stream>>>(x, Wk, kbuf);

  for (int iter = 0; iter < 3; ++iter) {
    seg_init_kernel<<<(NN * H + 255) / 256, 256, 0, stream>>>(segmax, segsum);
    ln_alpha_kernel<<<EH / 256, 256, 0, stream>>>(slots32, weff, lnS_g, lnS_b,
                                                  eidx, alpha, segmax);
    exp_sum_kernel<<<EH / 256, 256, 0, stream>>>(alpha, eidx, segmax, segsum);
    updates_kernel<<<EH / 256, 256, 0, stream>>>(alpha, eidx, segsum, kbuf, upd16);
    gru_wmma_kernel<<<dim3(HD / 16, E / 128), 256, 0, stream>>>(
        upd16, slots16, slots32, wih16, whh16, bih, bhh, stmp);
    float* outPtr = (iter == 2) ? (float*)d_out : slots32;
    mlp_wmma_kernel<<<EH / 64, 128, 0, stream>>>(
        stmp, lnM_g, lnM_b, w1_16, fc1b, w2_16, fc2b, outPtr, slots16);
  }
}
